// GroundTrans_34677565948527
// MI455X (gfx1250) — compile-verified, matrix operand order checked by reading, not run on hardware
//
#include <hip/hip_runtime.h>

// ---- CDNA5 WMMA types -------------------------------------------------------
typedef __attribute__((ext_vector_type(16))) __bf16        v16bf;
typedef __attribute__((ext_vector_type(8)))  float         v8f;
typedef __attribute__((ext_vector_type(8)))  unsigned short us8;
typedef __attribute__((ext_vector_type(16))) unsigned short us16;

union FragU { us16 u; v16bf b; us8 h[2]; };

__device__ __forceinline__ unsigned short f2bf(float f) {
  union { __bf16 h; unsigned short u; } c; c.h = (__bf16)f;   // HW RNE convert
  return c.u;
}
__device__ __forceinline__ unsigned int f2bf2(float a, float b) {
  union { __bf16 h[2]; unsigned int u; } c;                   // packed convert
  c.h[0] = (__bf16)a; c.h[1] = (__bf16)b;
  return c.u;
}

// ---- Generic tiled bf16-WMMA GEMM:  D = alpha * (A @ B) + bias --------------
// A: [M,K] row-major (fp32 or bf16), B: [K,N] row-major (fp32 or bf16)
// D: [M,N] row-major, or [N,M] if TR_OUT.  grid = (N/128, M/128, batch)
template<bool TA_BF, bool TB_BF, bool TD_BF, bool TR_OUT, bool STATS>
__global__ __launch_bounds__(256) void gt_gemm(
    const void* __restrict__ Ag, long long strideA, int lda,
    const void* __restrict__ Bg, long long strideB, int ldb,
    void*       __restrict__ Dg, long long strideD, int ldd,
    const float* __restrict__ bias, float alpha, int K,
    float* __restrict__ part)
{
  constexpr int LD = 40;                 // padded LDS row stride (ushorts)
  __shared__ unsigned short sA[128 * LD];
  __shared__ unsigned short sB[128 * LD];
  __shared__ float red[512];

  const int tid  = threadIdx.x;
  const int lane = tid & 31;
  const int wid  = tid >> 5;
  const int wm   = wid >> 1;             // 0..3 -> 32-row slab
  const int wn   = wid & 1;              // 0..1 -> 64-col slab
  const int gm0  = blockIdx.y * 128;
  const int gn0  = blockIdx.x * 128;
  const int z    = blockIdx.z;

  const float*          Af = nullptr; const unsigned short* Ab = nullptr;
  const float*          Bf = nullptr; const unsigned short* Bb = nullptr;
  if (TA_BF) Ab = (const unsigned short*)Ag + (long long)z * strideA + (long long)gm0 * lda;
  else       Af = (const float*)Ag          + (long long)z * strideA + (long long)gm0 * lda;
  if (TB_BF) Bb = (const unsigned short*)Bg + (long long)z * strideB + gn0;
  else       Bf = (const float*)Bg          + (long long)z * strideB + gn0;

  v8f acc[2][4];
  #pragma unroll
  for (int i = 0; i < 2; ++i)
    #pragma unroll
    for (int j = 0; j < 4; ++j) acc[i][j] = v8f{};

  const int r    = lane & 15;
  const int koff = (lane >> 4) * 8;

  for (int k0 = 0; k0 < K; k0 += 32) {
    __syncthreads();
    // ---- prefetch next K-chunk (global_prefetch_b8) ------------------------
    if (k0 + 32 < K) {
      if (tid < 128) {
        const char* pa = TA_BF ? (const char*)(Ab + (long long)tid * lda + k0 + 32)
                               : (const char*)(Af + (long long)tid * lda + k0 + 32);
        __builtin_prefetch(pa, 0, 3);
      } else if (tid < 160) {
        const int k = tid - 128;
        const char* pb = TB_BF ? (const char*)(Bb + (long long)(k0 + 32 + k) * ldb)
                               : (const char*)(Bf + (long long)(k0 + 32 + k) * ldb);
        __builtin_prefetch(pb, 0, 3);
      }
    }
    // ---- stage A tile [128 x 32] -> LDS bf16 -------------------------------
    if (TA_BF) {
      for (int i = tid; i < 512; i += 256) {
        int row = i >> 2, c8 = (i & 3) * 8;
        *(us8*)&sA[row * LD + c8] = *(const us8*)(Ab + (long long)row * lda + k0 + c8);
      }
    } else {
      for (int i = tid; i < 1024; i += 256) {
        int row = i >> 3, c4 = (i & 7) * 4;
        float4 v = *(const float4*)(Af + (long long)row * lda + k0 + c4);
        *(unsigned int*)&sA[row * LD + c4]     = f2bf2(v.x, v.y);
        *(unsigned int*)&sA[row * LD + c4 + 2] = f2bf2(v.z, v.w);
      }
    }
    // ---- stage B tile [32 x 128] -> LDS transposed bf16 --------------------
    if (TB_BF) {
      for (int i = tid; i < 512; i += 256) {
        int k = i >> 4, n = (i & 15) * 8;
        us8 v = *(const us8*)(Bb + (long long)(k0 + k) * ldb + n);
        #pragma unroll
        for (int j = 0; j < 8; ++j) sB[(n + j) * LD + k] = v[j];
      }
    } else {
      for (int i = tid; i < 1024; i += 256) {
        int k = i >> 5, n = (i & 31) * 4;
        float4 v = *(const float4*)(Bf + (long long)(k0 + k) * ldb + n);
        sB[(n + 0) * LD + k] = f2bf(v.x);
        sB[(n + 1) * LD + k] = f2bf(v.y);
        sB[(n + 2) * LD + k] = f2bf(v.z);
        sB[(n + 3) * LD + k] = f2bf(v.w);
      }
    }
    __syncthreads();

    // ---- fragments per ISA layout (16-bit A 16x32 / B 32x16) ---------------
    FragU af[2], bb[4];
    #pragma unroll
    for (int mt = 0; mt < 2; ++mt) {
      int row = wm * 32 + mt * 16 + r;
      af[mt].h[0] = *(const us8*)&sA[row * LD + koff];
      af[mt].h[1] = *(const us8*)&sA[row * LD + 16 + koff];
    }
    #pragma unroll
    for (int nt = 0; nt < 4; ++nt) {
      int col = wn * 64 + nt * 16 + r;
      bb[nt].h[0] = *(const us8*)&sB[col * LD + koff];
      bb[nt].h[1] = *(const us8*)&sB[col * LD + 16 + koff];
    }
    #pragma unroll
    for (int mt = 0; mt < 2; ++mt)
      #pragma unroll
      for (int nt = 0; nt < 4; ++nt)
        acc[mt][nt] = __builtin_amdgcn_wmma_f32_16x16x32_bf16(
            false, af[mt].b, false, bb[nt].b, (short)0, acc[mt][nt], false, false);
  }

  // ---- epilogue: hoisted bias, strength-reduced stores, optional stats -----
  float s1 = 0.f, s2 = 0.f;
  const int mb = (lane >> 4) * 8;        // C/D: lanes>=16 hold M = 8+v
  float bval[2][8];
  #pragma unroll
  for (int mt = 0; mt < 2; ++mt)
    #pragma unroll
    for (int v = 0; v < 8; ++v)
      bval[mt][v] = bias ? bias[gm0 + wm * 32 + mt * 16 + mb + v] : 0.0f;

  float*          Df = nullptr; unsigned short* Db = nullptr;
  if (TD_BF) Db = (unsigned short*)Dg + (long long)z * strideD;
  else       Df = (float*)Dg          + (long long)z * strideD;

  #pragma unroll
  for (int mt = 0; mt < 2; ++mt) {
    #pragma unroll
    for (int nt = 0; nt < 4; ++nt) {
      const int gmt = gm0 + wm * 32 + mt * 16 + mb;
      const int gn  = gn0 + wn * 64 + nt * 16 + r;
      float val[8];
      #pragma unroll
      for (int v = 0; v < 8; ++v) {
        val[v] = acc[mt][nt][v] * alpha + bval[mt][v];
        if (STATS) { s1 += val[v]; s2 += val[v] * val[v]; }
      }
      if (TR_OUT) {                       // consecutive in m -> vector stores
        const long long base = (long long)gn * ldd + gmt;
        if (TD_BF) {
          us8 o;
          #pragma unroll
          for (int v = 0; v < 8; ++v) o[v] = f2bf(val[v]);
          *(us8*)(Db + base) = o;         // one 16B store
        } else {
          float4 o0 = {val[0], val[1], val[2], val[3]};
          float4 o1 = {val[4], val[5], val[6], val[7]};
          *(float4*)(Df + base)     = o0;
          *(float4*)(Df + base + 4) = o1;
        }
      } else {                            // strided by ldd
        long long base = (long long)gmt * ldd + gn;
        #pragma unroll
        for (int v = 0; v < 8; ++v) {
          if (TD_BF) Db[base] = f2bf(val[v]); else Df[base] = val[v];
          base += ldd;
        }
      }
    }
  }

  if (STATS) {  // deterministic in-block tree reduction -> per-block partials
    red[tid] = s1; red[256 + tid] = s2;
    __syncthreads();
    for (int st = 128; st > 0; st >>= 1) {
      if (tid < st) { red[tid] += red[tid + st]; red[256 + tid] += red[256 + tid + st]; }
      __syncthreads();
    }
    if (tid == 0) {
      int blk = (z * gridDim.y + blockIdx.y) * gridDim.x + blockIdx.x;
      part[blk * 2 + 0] = red[0];
      part[blk * 2 + 1] = red[256];
    }
  }
}

// ---- reduce per-block partials (64 blocks per batch, fixed order) ----------
__global__ void gt_redstats(const float* __restrict__ part, float* __restrict__ stats) {
  int b = threadIdx.x;
  if (b < 8) {
    float s1 = 0.f, s2 = 0.f;
    for (int i = 0; i < 64; ++i) { s1 += part[(b * 64 + i) * 2]; s2 += part[(b * 64 + i) * 2 + 1]; }
    stats[2 * b] = s1; stats[2 * b + 1] = s2;
  }
}

// ---- GroupNorm(1 group) over (C,H,W) per sample, fused scale/shift ---------
__global__ __launch_bounds__(256) void gt_norm(
    float* __restrict__ zbuf, const float* __restrict__ stats,
    const float* __restrict__ gamma, const float* __restrict__ beta)
{
  const int idx = blockIdx.x * 256 + threadIdx.x;  // float4 index, 2,097,152 total
  const int b   = idx >> 18;                       // / (256*4096/4)
  const int rem = idx & 262143;
  const int c   = rem >> 10;                       // / (4096/4)
  const float inv = 1.0f / 1048576.0f;             // 1/(C*Hl*Wl)
  const float mu  = stats[2 * b] * inv;
  const float var = stats[2 * b + 1] * inv - mu * mu;
  const float rs  = rsqrtf(var + 1e-5f);
  const float g   = gamma[c] * rs;
  const float o   = beta[c] - mu * g;
  float4* zp = (float4*)zbuf;
  float4 v = zp[idx];
  v.x = v.x * g + o; v.y = v.y * g + o; v.z = v.z * g + o; v.w = v.w * g + o;
  zp[idx] = v;
}

// ---- driver -----------------------------------------------------------------
extern "C" void kernel_launch(void* const* d_in, const int* in_sizes, int n_in,
                              void* d_out, int out_size, void* d_ws, size_t ws_size,
                              hipStream_t stream) {
  (void)in_sizes; (void)n_in; (void)out_size; (void)ws_size;
  const float* x_high = (const float*)d_in[0];
  const float* x_low  = (const float*)d_in[1];
  const float* Wg = (const float*)d_in[2];
  const float* bg = (const float*)d_in[3];
  const float* Wt = (const float*)d_in[4];
  const float* bt = (const float*)d_in[5];
  const float* Wp = (const float*)d_in[6];
  const float* bp = (const float*)d_in[7];
  const float* Wz = (const float*)d_in[8];
  const float* bz = (const float*)d_in[9];
  const float* gamma = (const float*)d_in[10];
  const float* beta  = (const float*)d_in[11];
  float* out = (float*)d_out;

  // workspace layout (256B-aligned)
  char* ws = (char*)d_ws;
  float*          stats = (float*)(ws + 0);          // 16 f32
  float*          part  = (float*)(ws + 256);        // 512*2 f32
  unsigned short* P     = (unsigned short*)(ws + 8192);      // phi    [B,128,1024] bf16
  unsigned short* Gt    = (unsigned short*)(ws + 2105344);   // g^T    [B,1024,128] bf16
  unsigned short* Mt    = (unsigned short*)(ws + 4202496);   // M^T/Nh [B,128,128]  bf16
  float*          Wze   = (float*)(ws + 4464640);            // Wz_eff [B,256,128]  f32
  unsigned short* Th    = (unsigned short*)(ws + 5513216);   // theta  [B,128,4096] bf16

  dim3 blk(256);
  // g^T = (Wg@xh + bg)^T          [Nh,Ci]   (M=128,N=1024,K=256)
  gt_gemm<false,false,true,true, false><<<dim3(8,1,8),  blk, 0, stream>>>(
      Wg, 0, 256, x_high, 262144, 1024, Gt, 131072, 128, bg, 1.0f, 256, nullptr);
  // phi = Wp@xh + bp              [Ci,Nh]
  gt_gemm<false,false,true,false,false><<<dim3(8,1,8),  blk, 0, stream>>>(
      Wp, 0, 256, x_high, 262144, 1024, P, 131072, 1024, bp, 1.0f, 256, nullptr);
  // M^T/Nh = (phi @ g^T)^T / 1024 [Ci,Ci]   (K=1024)
  gt_gemm<true, true, true,true, false><<<dim3(1,1,8),  blk, 0, stream>>>(
      P, 131072, 1024, Gt, 131072, 128, Mt, 16384, 128, nullptr, 1.0f/1024.0f, 1024, nullptr);
  // Wz_eff = Wz @ (M^T/Nh)        [C,Ci]    (M=256,N=128,K=128)
  gt_gemm<false,true, false,false,false><<<dim3(1,2,8), blk, 0, stream>>>(
      Wz, 0, 128, Mt, 16384, 128, Wze, 32768, 128, nullptr, 1.0f, 128, nullptr);
  // theta = Wt@xl + bt            [Ci,Nl]   (M=128,N=4096,K=256)
  gt_gemm<false,false,true,false,false><<<dim3(32,1,8), blk, 0, stream>>>(
      Wt, 0, 256, x_low, 1048576, 4096, Th, 524288, 4096, bt, 1.0f, 256, nullptr);
  // z = Wz_eff @ theta + bz       [C,Nl]    (M=256,N=4096,K=128) + stats
  gt_gemm<false,true, false,false,true ><<<dim3(32,2,8), blk, 0, stream>>>(
      Wze, 32768, 128, Th, 524288, 4096, out, 1048576, 4096, bz, 1.0f, 128, part);
  gt_redstats<<<1, 64, 0, stream>>>(part, stats);
  gt_norm<<<dim3(8192), blk, 0, stream>>>(out, stats, gamma, beta);
}
